// ScaledDotProductAttention_42520176230534
// MI455X (gfx1250) — matrix operand order repaired and checked by reading, not compiled, not run on hardware
//
#include <hip/hip_runtime.h>
#include <stdint.h>

typedef __attribute__((ext_vector_type(16))) __bf16 v16bf;
typedef __attribute__((ext_vector_type(8)))  __bf16 v8bf;
typedef __attribute__((ext_vector_type(4)))  __bf16 v4bf;
typedef __attribute__((ext_vector_type(8)))  float  v8f;

namespace {
constexpr int B = 4, H = 16, S = 1024, D = 64;
constexpr int BQ = 128;   // query rows per workgroup (8 waves x 16)
constexpr int KT = 64;    // key tile size
constexpr int LP = 80;    // LDS row pitch in bf16 elems (pad, keeps 16B alignment)
constexpr int TILEB = KT * D * 4;          // bytes per f32 staging tile (16 KB)
constexpr float INV_T   = 0.125f;          // 1 / sqrt(d_head=64)
constexpr float NEG_INF = -1.0e9f;
}

__device__ __forceinline__ v16bf cat8(v8bf a, v8bf b) {
  v16bf r;
#pragma unroll
  for (int i = 0; i < 8; i++) { r[i] = a[i]; r[i + 8] = b[i]; }
  return r;
}

// B-operand fragment: K = {start..start+7, start+16..start+23} from one LDS row.
// (16-bit B 32x16: v0-3 K = 8*hi.., v4-7 K = 16+8*hi..)
__device__ __forceinline__ v16bf lds_bfrag(const __bf16* rowbase, int start) {
  v8bf a = *(const v8bf*)(rowbase + start);
  v8bf b = *(const v8bf*)(rowbase + start + 16);
  return cat8(a, b);
}

// A-operand fragment from LDS bf16 row: K = {lo..lo+7, hb..hb+7}
// (16-bit A 16x32: v0-3 K = 8*hi.., v4-7 K = 16+8*hi..)
__device__ __forceinline__ v16bf lds_afrag(const __bf16* rowbase, int lo, int hb) {
  v8bf a = *(const v8bf*)(rowbase + lo);
  v8bf b = *(const v8bf*)(rowbase + hb);
  return cat8(a, b);
}

// A-operand fragment from a global f32 row, scaled and converted to bf16.
__device__ __forceinline__ v16bf global_afrag(const float* p, int lo, int hb, float sc) {
  const float4 x0 = *(const float4*)(p + lo);
  const float4 x1 = *(const float4*)(p + lo + 4);
  const float4 y0 = *(const float4*)(p + hb);
  const float4 y1 = *(const float4*)(p + hb + 4);
  float lof[8] = {x0.x, x0.y, x0.z, x0.w, x1.x, x1.y, x1.z, x1.w};
  float hbf[8] = {y0.x, y0.y, y0.z, y0.w, y1.x, y1.y, y1.z, y1.w};
  v16bf r;
#pragma unroll
  for (int i = 0; i < 8; i++) {
    r[i]     = (__bf16)(lof[i] * sc);
    r[i + 8] = (__bf16)(hbf[i] * sc);
  }
  return r;
}

// Async copy of 64 bytes (4x B128) global -> LDS, per lane. ASYNCcnt-tracked.
// Same INST_OFFSET is applied to both the LDS and global addresses (ISA 8/§4.4).
__device__ __forceinline__ void async_cp64(uint32_t lds, uint64_t gaddr) {
  asm volatile(
      "global_load_async_to_lds_b128 %0, %1, off\n\t"
      "global_load_async_to_lds_b128 %0, %1, off offset:16\n\t"
      "global_load_async_to_lds_b128 %0, %1, off offset:32\n\t"
      "global_load_async_to_lds_b128 %0, %1, off offset:48"
      :: "v"(lds), "v"(gaddr) : "memory");
}

__device__ __forceinline__ void wait_async0() {
  asm volatile("s_wait_asynccnt 0x0" ::: "memory");
}

__global__ __launch_bounds__(256, 1)
void fattn_dual_kernel(const float* __restrict__ q,  const float* __restrict__ k,
                       const float* __restrict__ v,  const float* __restrict__ qu,
                       const float* __restrict__ ku, const float* __restrict__ theta,
                       const int* __restrict__ mask, float* __restrict__ out) {
  // f32 double-buffered async staging (96 KB)
  __shared__ __align__(16) float Ks [2][KT * D];
  __shared__ __align__(16) float KUs[2][KT * D];
  __shared__ __align__(16) float Vs [2][KT * D];
  // bf16 WMMA tiles (30 KB) + per-wave P tiles (20 KB)
  __shared__ __align__(16) __bf16 Kl [KT * LP];       // K tile, [key][d]
  __shared__ __align__(16) __bf16 KUl[KT * LP];       // K_u tile, [key][d]
  __shared__ __align__(16) __bf16 VTl[D  * LP];       // V tile transposed, [d][key]
  __shared__ __align__(16) __bf16 Pl [8 * 16 * LP];   // per-wave P tiles, [q][key]

  const int tid  = threadIdx.x;
  const int wave = tid >> 5;
  const int lane = tid & 31;
  const int hi   = lane >> 4;
  const int ln   = lane & 15;

  const int gid = blockIdx.x;
  const int qb  = gid & 7;          // S/BQ = 8
  const int h   = (gid >> 3) & 15;  // H = 16
  const int b   = gid >> 7;
  const int q0  = qb * BQ;
  const int qw  = q0 + wave * 16;   // this wave's first q row

  const size_t bh = (size_t)(b * H + h) * S * D;

  // ---- per-thread async staging addresses (64-byte slice per thread) ----
  const uint64_t gk  = (uint64_t)(uintptr_t)(k  + bh) + (uint64_t)tid * 64;
  const uint64_t gku = (uint64_t)(uintptr_t)(ku + bh) + (uint64_t)tid * 64;
  const uint64_t gv  = (uint64_t)(uintptr_t)(v  + bh) + (uint64_t)tid * 64;
  uint32_t ks_l[2], kus_l[2], vs_l[2];
#pragma unroll
  for (int i = 0; i < 2; i++) {
    ks_l[i]  = (uint32_t)(uintptr_t)(&Ks [i][0]) + (uint32_t)tid * 64;
    kus_l[i] = (uint32_t)(uintptr_t)(&KUs[i][0]) + (uint32_t)tid * 64;
    vs_l[i]  = (uint32_t)(uintptr_t)(&Vs [i][0]) + (uint32_t)tid * 64;
  }

  // ---- Q / Q_u A-fragments (persistent, pre-scaled by 1/T) ----
  const float* qp  = q  + bh + (size_t)(qw + ln) * D;
  const float* qup = qu + bh + (size_t)(qw + ln) * D;
  v16bf qa[2], qua[2];
#pragma unroll
  for (int c = 0; c < 2; c++) {
    const int lo = c * 32 + hi * 8;
    const int hb = c * 32 + 16 + hi * 8;
    qa[c]  = global_afrag(qp,  lo, hb, INV_T);
    qua[c] = global_afrag(qup, lo, hb, INV_T);
  }

  // per-lane bases for theta [1,H,S,S] and mask [B,1,S,S] in C-layout
  const size_t tbase = (size_t)h * S * S + (size_t)(qw + 8 * hi) * S + ln;
  const size_t mbase = (size_t)b * S * S + (size_t)(qw + 8 * hi) * S + ln;

  float m_run[8], l_run[8];
  v8f acc[4];
#pragma unroll
  for (int r = 0; r < 8; r++) { m_run[r] = -INFINITY; l_run[r] = 0.0f; }
#pragma unroll
  for (int dt = 0; dt < 4; dt++) {
#pragma unroll
    for (int r = 0; r < 8; r++) acc[dt][r] = 0.0f;
  }

  // ---- prologue: async-issue key tile 0 into buffer 0 ----
  async_cp64(ks_l[0],  gk);
  async_cp64(kus_l[0], gku);
  async_cp64(vs_l[0],  gv);

  for (int kt = 0; kt < S / KT; kt++) {
    const int k0  = kt * KT;
    const int cur = kt & 1;

    wait_async0();      // this wave's slice of tile kt has landed in LDS
    __syncthreads();    // all slices visible; previous compute's LDS reads done

    // overlap: issue tile kt+1 into the other staging buffer
    if (kt + 1 < S / KT) {
      const uint64_t o = (uint64_t)(kt + 1) * TILEB;
      async_cp64(ks_l[cur ^ 1],  gk  + o);
      async_cp64(kus_l[cur ^ 1], gku + o);
      async_cp64(vs_l[cur ^ 1],  gv  + o);
    }

    // ---- convert pass: LDS f32 stage -> bf16 WMMA tiles (V transposed) ----
#pragma unroll
    for (int i = 0; i < 4; i++) {
      const int f  = tid + i * 256;     // float4 index, KT*D/4 = 1024 total
      const int kr = f >> 4;            // key row 0..63
      const int cc = (f & 15) * 4;      // d col 0..60
      const float4 kv  = *(const float4*)&Ks [cur][f * 4];
      const float4 kuv = *(const float4*)&KUs[cur][f * 4];
      const float4 vv  = *(const float4*)&Vs [cur][f * 4];
      v4bf kb, kub;
      kb[0]  = (__bf16)kv.x;  kb[1]  = (__bf16)kv.y;  kb[2]  = (__bf16)kv.z;  kb[3]  = (__bf16)kv.w;
      kub[0] = (__bf16)kuv.x; kub[1] = (__bf16)kuv.y; kub[2] = (__bf16)kuv.z; kub[3] = (__bf16)kuv.w;
      *(v4bf*)&Kl [kr * LP + cc] = kb;
      *(v4bf*)&KUl[kr * LP + cc] = kub;
      VTl[(cc + 0) * LP + kr] = (__bf16)vv.x;
      VTl[(cc + 1) * LP + kr] = (__bf16)vv.y;
      VTl[(cc + 2) * LP + kr] = (__bf16)vv.z;
      VTl[(cc + 3) * LP + kr] = (__bf16)vv.w;
    }
    __syncthreads();

    // ---- scores for 4 sub-tiles of 16 keys: dual-term WMMA + theta + mask ----
    float sc_[4][8];
    float tmax[8];
#pragma unroll
    for (int r = 0; r < 8; r++) tmax[r] = -INFINITY;

#pragma unroll
    for (int t = 0; t < 4; t++) {
      v8f s;
#pragma unroll
      for (int r = 0; r < 8; r++) s[r] = 0.0f;
      const __bf16* kr0  = &Kl [(t * 16 + ln) * LP];
      const __bf16* kur0 = &KUl[(t * 16 + ln) * LP];
#pragma unroll
      for (int c = 0; c < 2; c++) {
        const int bs = c * 32 + hi * 8;
        v16bf kb = lds_bfrag(kr0, bs);
        s = __builtin_amdgcn_wmma_f32_16x16x32_bf16(false, qa[c],  false, kb,  (short)0, s, false, false);
        v16bf kub = lds_bfrag(kur0, bs);
        s = __builtin_amdgcn_wmma_f32_16x16x32_bf16(false, qua[c], false, kub, (short)0, s, false, false);
      }
#pragma unroll
      for (int r = 0; r < 8; r++) {
        const size_t off = (size_t)r * S + (size_t)(k0 + t * 16);
        float x = s[r] + theta[tbase + off];
        x = (mask[mbase + off] != 0) ? x : NEG_INF;
        sc_[t][r] = x;
        tmax[r] = fmaxf(tmax[r], x);
      }
    }

    // ---- online softmax: row max across 16 lanes, rescale state ----
#pragma unroll
    for (int r = 0; r < 8; r++) {
      float m = tmax[r];
      m = fmaxf(m, __shfl_xor(m, 1, 32));
      m = fmaxf(m, __shfl_xor(m, 2, 32));
      m = fmaxf(m, __shfl_xor(m, 4, 32));
      m = fmaxf(m, __shfl_xor(m, 8, 32));
      const float mn   = fmaxf(m_run[r], m);
      const float corr = __expf(m_run[r] - mn);
      m_run[r] = mn;
      l_run[r] *= corr;
#pragma unroll
      for (int dt = 0; dt < 4; dt++) acc[dt][r] *= corr;
    }

    // ---- P = exp(s - m), row sums, bounce P through per-wave LDS tile ----
    float rsum[8];
#pragma unroll
    for (int r = 0; r < 8; r++) rsum[r] = 0.0f;
#pragma unroll
    for (int t = 0; t < 4; t++) {
#pragma unroll
      for (int r = 0; r < 8; r++) {
        const float p = __expf(sc_[t][r] - m_run[r]);
        rsum[r] += p;
        Pl[(wave * 16 + r + 8 * hi) * LP + t * 16 + ln] = (__bf16)p;
      }
    }
#pragma unroll
    for (int r = 0; r < 8; r++) {
      float sm = rsum[r];
      sm += __shfl_xor(sm, 1, 32);
      sm += __shfl_xor(sm, 2, 32);
      sm += __shfl_xor(sm, 4, 32);
      sm += __shfl_xor(sm, 8, 32);
      l_run[r] += sm;
    }

    // ---- PV: acc[dt] += P(16xKT) * V(KTx16dt), A from LDS, B from V^T LDS ----
    const __bf16* prow = &Pl[(wave * 16 + ln) * LP];
#pragma unroll
    for (int c = 0; c < 2; c++) {
      const int lo = c * 32 + hi * 8;
      const int hb = c * 32 + 16 + hi * 8;
      v16bf pA = lds_afrag(prow, lo, hb);
      const int bs = c * 32 + hi * 8;
#pragma unroll
      for (int dt = 0; dt < 4; dt++) {
        v16bf vB = lds_bfrag(&VTl[(dt * 16 + ln) * LP], bs);
        acc[dt] = __builtin_amdgcn_wmma_f32_16x16x32_bf16(false, pA, false, vB, (short)0, acc[dt], false, false);
      }
    }
  }

  // ---- epilogue: normalize and store f32 output ----
  float* op = out + bh;
#pragma unroll
  for (int r = 0; r < 8; r++) {
    const float inv = 1.0f / l_run[r];
#pragma unroll
    for (int dt = 0; dt < 4; dt++) {
      op[(size_t)(qw + r + 8 * hi) * D + dt * 16 + ln] = acc[dt][r] * inv;
    }
  }
}

extern "C" void kernel_launch(void* const* d_in, const int* in_sizes, int n_in,
                              void* d_out, int out_size, void* d_ws, size_t ws_size,
                              hipStream_t stream) {
  const float* q     = (const float*)d_in[0];
  const float* k     = (const float*)d_in[1];
  const float* v     = (const float*)d_in[2];
  const float* qu    = (const float*)d_in[3];
  const float* ku    = (const float*)d_in[4];
  const float* theta = (const float*)d_in[5];
  const int*   mask  = (const int*)d_in[6];
  float* out = (float*)d_out;

  dim3 grid(B * H * (S / BQ));   // 4*16*8 = 512 workgroups
  dim3 block(256);               // 8 waves (wave32)
  fattn_dual_kernel<<<grid, block, 0, stream>>>(q, k, v, qu, ku, theta, mask, out);
}
